// GraphAttentionNetwork_74448963108868
// MI455X (gfx1250) — compile-verified, matrix operand order checked by reading, not run on hardware
//
#include <hip/hip_runtime.h>
#include <hip/hip_bf16.h>

typedef __attribute__((ext_vector_type(16))) __bf16 v16bf;
typedef __attribute__((ext_vector_type(8)))  float  v8f;

#define NN 100000
#define EE 1600000

union ABfrag { v16bf v; unsigned u[8]; };

__device__ __forceinline__ unsigned short f32_to_bf16(float f) {
    unsigned u = __float_as_uint(f);
    u += 0x7FFFu + ((u >> 16) & 1u);
    return (unsigned short)(u >> 16);
}

// ---------- generic fill ----------
__global__ void fill_u32(unsigned* __restrict__ p, unsigned v, long long n) {
    long long i = (long long)blockIdx.x * blockDim.x + threadIdx.x;
    if (i < n) p[i] = v;
}

// ---------- cast fp32 -> bf16 (layer-0 input) ----------
__global__ void cast_f32_bf16(const float* __restrict__ src, unsigned short* __restrict__ dst,
                              long long n) {
    long long i = (long long)blockIdx.x * blockDim.x + threadIdx.x;
    if (i < n) dst[i] = f32_to_bf16(src[i]);
}

// ---------- transpose + cast weights: W[k][o] (fp32) -> Wt[o][k] (bf16) ----------
__global__ void cast_wt(const float* __restrict__ W, unsigned short* __restrict__ Wt, int K, int O) {
    int i = blockIdx.x * blockDim.x + threadIdx.x;
    if (i >= K * O) return;
    int k = i / O, o = i % O;
    Wt[(size_t)o * K + k] = f32_to_bf16(W[i]);
}

// ---------- fragment load: row-major bf16, 16-bit A/B WMMA layout ----------
// lanes 0-15 hold K {0..7, 16..23}; lanes 16-31 hold K {8..15, 24..31}
__device__ __forceinline__ void load_frag(const unsigned short* __restrict__ rowbase,
                                          int lhalf, ABfrag& f) {
    const uint4 c0 = *(const uint4*)(rowbase + lhalf * 8);
    const uint4 c1 = *(const uint4*)(rowbase + 16 + lhalf * 8);
    f.u[0] = c0.x; f.u[1] = c0.y; f.u[2] = c0.z; f.u[3] = c0.w;
    f.u[4] = c1.x; f.u[5] = c1.y; f.u[6] = c1.z; f.u[7] = c1.w;
}

// ---------- GEMM: Y[M][O] (fp32) = Xb[M][K] (bf16) * W[K][O] via Wt[O][K] (bf16) ----------
// block = 256 thr (8 waves). wave -> 32x32 tile via 2x2 v_wmma_f32_16x16x32_bf16.
__global__ __launch_bounds__(256) void gemm_wmma_bf16(
        const unsigned short* __restrict__ Xb, const unsigned short* __restrict__ Wt,
        float* __restrict__ Y, int M, int K, int O) {
    const int tid   = threadIdx.x;
    const int wave  = tid >> 5;
    const int lane  = tid & 31;
    const int lrow  = lane & 15;
    const int lhalf = lane >> 4;
    const int wm = wave & 3;       // 4 waves along M
    const int wn = wave >> 2;      // 2 waves along N
    const int tm = blockIdx.y * 128 + wm * 32;
    const int tn = blockIdx.x * 64  + wn * 32;

    int arow0 = tm + lrow;      if (arow0 >= M) arow0 = M - 1;
    int arow1 = tm + 16 + lrow; if (arow1 >= M) arow1 = M - 1;
    const unsigned short* aptr0 = Xb + (size_t)arow0 * K;
    const unsigned short* aptr1 = Xb + (size_t)arow1 * K;
    const unsigned short* bptr0 = Wt + (size_t)(tn + lrow) * K;
    const unsigned short* bptr1 = Wt + (size_t)(tn + 16 + lrow) * K;

    v8f acc[2][2];
#pragma unroll
    for (int i = 0; i < 2; ++i)
#pragma unroll
        for (int j = 0; j < 2; ++j)
            acc[i][j] = v8f{0.f,0.f,0.f,0.f,0.f,0.f,0.f,0.f};

    for (int k0 = 0; k0 < K; k0 += 32) {
        ABfrag a0, a1, b0, b1;
        load_frag(aptr0 + k0, lhalf, a0);
        load_frag(aptr1 + k0, lhalf, a1);
        load_frag(bptr0 + k0, lhalf, b0);
        load_frag(bptr1 + k0, lhalf, b1);
        if (k0 + 32 < K) {
            __builtin_prefetch(aptr0 + k0 + 32, 0, 1);
            __builtin_prefetch(aptr1 + k0 + 32, 0, 1);
        }
        acc[0][0] = __builtin_amdgcn_wmma_f32_16x16x32_bf16(false, a0.v, false, b0.v, (short)0, acc[0][0], false, false);
        acc[0][1] = __builtin_amdgcn_wmma_f32_16x16x32_bf16(false, a0.v, false, b1.v, (short)0, acc[0][1], false, false);
        acc[1][0] = __builtin_amdgcn_wmma_f32_16x16x32_bf16(false, a1.v, false, b0.v, (short)0, acc[1][0], false, false);
        acc[1][1] = __builtin_amdgcn_wmma_f32_16x16x32_bf16(false, a1.v, false, b1.v, (short)0, acc[1][1], false, false);
    }

#pragma unroll
    for (int i = 0; i < 2; ++i)
#pragma unroll
        for (int j = 0; j < 2; ++j)
#pragma unroll
            for (int r = 0; r < 8; ++r) {
                int row = tm + i * 16 + lhalf * 8 + r;
                int col = tn + j * 16 + lrow;
                if (row < M) Y[(size_t)row * O + col] = acc[i][j][r];
            }
}

// ---------- attention logits: one wave per (node, head), C=64 per head always ----------
__global__ void attn_logits(const float* __restrict__ hbuf,
                            const float* __restrict__ asrc, const float* __restrict__ adst,
                            float* __restrict__ al_s, float* __restrict__ al_d,
                            int nnodes, int H) {
    int wv = blockIdx.x * (blockDim.x >> 5) + (threadIdx.x >> 5);
    int lane = threadIdx.x & 31;
    if (wv >= nnodes * H) return;
    int n = wv / H, h = wv % H;
    float2 hv = *(const float2*)(hbuf + (size_t)n * H * 64 + h * 64 + lane * 2);
    float2 as = *(const float2*)(asrc + h * 64 + lane * 2);
    float2 ad = *(const float2*)(adst + h * 64 + lane * 2);
    float ps = hv.x * as.x + hv.y * as.y;
    float pd = hv.x * ad.x + hv.y * ad.y;
#pragma unroll
    for (int off = 16; off; off >>= 1) {
        ps += __shfl_xor(ps, off, 32);
        pd += __shfl_xor(pd, off, 32);
    }
    if (lane == 0) { al_s[wv] = ps; al_d[wv] = pd; }
}

// ---------- edge logits + segment max (float atomic-max via int/uint trick) ----------
__global__ void edge_logits(const long long* __restrict__ src, const long long* __restrict__ dst,
                            const float* __restrict__ al_s, const float* __restrict__ al_d,
                            float* __restrict__ ebuf, unsigned* __restrict__ mmax,
                            long long nE, int H) {
    long long e = (long long)blockIdx.x * blockDim.x + threadIdx.x;
    if (e >= nE) return;
    int s = (int)src[e], d = (int)dst[e];
    for (int h = 0; h < H; ++h) {
        float v = al_s[s * H + h] + al_d[d * H + h];
        v = v > 0.f ? v : 0.2f * v;                       // leaky_relu(0.2)
        ebuf[e * H + h] = v;
        if (v >= 0.f) atomicMax((int*)(mmax + d * H + h), __float_as_int(v));
        else          atomicMin(mmax + d * H + h, (unsigned)__float_as_int(v));
    }
}

// ---------- m = isfinite(m) ? m : 0 ----------
__global__ void m_fix(float* __restrict__ m, long long n) {
    long long i = (long long)blockIdx.x * blockDim.x + threadIdx.x;
    if (i >= n) return;
    unsigned u = __float_as_uint(m[i]);
    if ((u & 0x7F800000u) == 0x7F800000u) m[i] = 0.f;
}

// ---------- p = exp(e - m[dst]); denom[dst] += p ----------
__global__ void edge_exp(const long long* __restrict__ dst,
                         float* __restrict__ ebuf, const float* __restrict__ m,
                         float* __restrict__ denom, long long nE, int H) {
    long long e = (long long)blockIdx.x * blockDim.x + threadIdx.x;
    if (e >= nE) return;
    int d = (int)dst[e];
    for (int h = 0; h < H; ++h) {
        float p = __expf(ebuf[e * H + h] - m[d * H + h]);
        ebuf[e * H + h] = p;
        atomicAdd(denom + d * H + h, p);
    }
}

// ---------- agg[dst] += h[src] * alpha  (HC/4 threads per edge, float4 gather) ----------
__global__ void agg_kernel(const long long* __restrict__ src, const long long* __restrict__ dst,
                           const float* __restrict__ hbuf, const float* __restrict__ p,
                           const float* __restrict__ denom, float* __restrict__ agg,
                           long long nE, int HC) {
    long long gid = (long long)blockIdx.x * blockDim.x + threadIdx.x;
    int tpe = HC >> 2;
    long long e = gid / tpe;
    if (e >= nE) return;
    int c4 = (int)(gid % tpe) * 4;
    int H = HC >> 6;
    int head = c4 >> 6;
    int s = (int)src[e], d = (int)dst[e];
    float alpha = p[e * H + head] / (denom[d * H + head] + 1e-16f);
    float4 hv = *(const float4*)(hbuf + (size_t)s * HC + c4);
    float* o = agg + (size_t)d * HC + c4;
    atomicAdd(o + 0, hv.x * alpha);
    atomicAdd(o + 1, hv.y * alpha);
    atomicAdd(o + 2, hv.z * alpha);
    atomicAdd(o + 3, hv.w * alpha);
}

// ---------- out = ELU(agg + bias) -> bf16 (feeds next layer's GEMM) ----------
__global__ void finalize_bf16(const float* __restrict__ agg, const float* __restrict__ bias,
                              unsigned short* __restrict__ out, long long total, int O) {
    long long i = (long long)blockIdx.x * blockDim.x + threadIdx.x;
    if (i >= total) return;
    float v = agg[i] + bias[i % O];
    v = v > 0.f ? v : (__expf(v) - 1.f);
    out[i] = f32_to_bf16(v);
}

// ---------- out = agg + bias -> fp32 (final layer, no ELU) ----------
__global__ void finalize_f32(const float* __restrict__ agg, const float* __restrict__ bias,
                             float* __restrict__ out, long long total, int O) {
    long long i = (long long)blockIdx.x * blockDim.x + threadIdx.x;
    if (i >= total) return;
    out[i] = agg[i] + bias[i % O];
}

extern "C" void kernel_launch(void* const* d_in, const int* in_sizes, int n_in,
                              void* d_out, int out_size, void* d_ws, size_t ws_size,
                              hipStream_t stream) {
    const float*     x    = (const float*)d_in[0];
    const long long* ei   = (const long long*)d_in[1];
    const long long* srcp = ei;
    const long long* dstp = ei + EE;
    const float* Wp[4]    = {(const float*)d_in[3],  (const float*)d_in[7],
                             (const float*)d_in[11], (const float*)d_in[15]};
    const float* ASRC[4]  = {(const float*)d_in[4],  (const float*)d_in[8],
                             (const float*)d_in[12], (const float*)d_in[16]};
    const float* ADST[4]  = {(const float*)d_in[5],  (const float*)d_in[9],
                             (const float*)d_in[13], (const float*)d_in[17]};
    const float* Bp[4]    = {(const float*)d_in[6],  (const float*)d_in[10],
                             (const float*)d_in[14], (const float*)d_in[18]};

    // ---- workspace carve-out (256B aligned) ----
    char*  base = (char*)d_ws;
    size_t off  = 0;
    auto carve = [&](size_t bytes) {
        void* p = base + off;
        off = (off + bytes + 255) & ~(size_t)255;
        return p;
    };
    float*          hbuf  = (float*)carve((size_t)NN * 256 * 4);          // GEMM output (fp32)
    float*          agg   = (float*)carve((size_t)NN * 256 * 4);          // aggregation accum
    unsigned short* xbf   = (unsigned short*)carve((size_t)NN * 256 * 2); // bf16 layer input
    float*          ebuf  = (float*)carve((size_t)EE * 4 * 4);            // edge logits / probs
    float*          al_s  = (float*)carve((size_t)NN * 4 * 4);
    float*          al_d  = (float*)carve((size_t)NN * 4 * 4);
    float*          mbuf  = (float*)carve((size_t)NN * 4 * 4);
    float*          denom = (float*)carve((size_t)NN * 4 * 4);
    unsigned short* wt    = (unsigned short*)carve((size_t)256 * 256 * 2);

    const int Kin[4] = {128, 256, 256, 256};
    const int Oc[4]  = {256, 256, 256, 64};
    const int Hh[4]  = {4, 4, 4, 1};

    // layer-0 input -> bf16 once
    cast_f32_bf16<<<((long long)NN * 128 + 255) / 256, 256, 0, stream>>>(x, xbf, (long long)NN * 128);

    for (int L = 0; L < 4; ++L) {
        const int K = Kin[L], O = Oc[L], H = Hh[L], HC = O; // HC == H*64 for every layer
        // 1) weights -> bf16 transposed
        cast_wt<<<(K * O + 255) / 256, 256, 0, stream>>>(Wp[L], wt, K, O);
        // 2) h = x @ W  (WMMA bf16, bf16 operands end-to-end)
        gemm_wmma_bf16<<<dim3(O / 64, (NN + 127) / 128), 256, 0, stream>>>(xbf, wt, hbuf, NN, K, O);
        // 3) clear per-layer state
        fill_u32<<<((long long)NN * H + 255) / 256, 256, 0, stream>>>((unsigned*)mbuf, 0xFF800000u, (long long)NN * H);
        fill_u32<<<((long long)NN * H + 255) / 256, 256, 0, stream>>>((unsigned*)denom, 0u, (long long)NN * H);
        fill_u32<<<((long long)NN * O + 255) / 256, 256, 0, stream>>>((unsigned*)agg, 0u, (long long)NN * O);
        // 4) attention logits
        attn_logits<<<(NN * H + 7) / 8, 256, 0, stream>>>(hbuf, ASRC[L], ADST[L], al_s, al_d, NN, H);
        // 5) edge logits + segment max
        edge_logits<<<(EE + 255) / 256, 256, 0, stream>>>(srcp, dstp, al_s, al_d, ebuf, (unsigned*)mbuf, EE, H);
        // 6) sanitize max
        m_fix<<<((long long)NN * H + 255) / 256, 256, 0, stream>>>(mbuf, (long long)NN * H);
        // 7) exp + denom
        edge_exp<<<(EE + 255) / 256, 256, 0, stream>>>(dstp, ebuf, mbuf, denom, EE, H);
        // 8) aggregation (scatter-add)
        long long tot = (long long)EE * (HC >> 2);
        agg_kernel<<<(tot + 255) / 256, 256, 0, stream>>>(srcp, dstp, hbuf, ebuf, denom, agg, EE, HC);
        // 9) bias + activation; layers 0-2 feed next GEMM as bf16, layer 3 -> fp32 d_out
        if (L < 3)
            finalize_bf16<<<((long long)NN * O + 255) / 256, 256, 0, stream>>>(agg, Bp[L], xbf, (long long)NN * O, O);
        else
            finalize_f32<<<((long long)NN * O + 255) / 256, 256, 0, stream>>>(agg, Bp[L], (float*)d_out, (long long)NN * O, O);
    }
}
